// LSTM_cell_31439160607293
// MI455X (gfx1250) — compile-verified, matrix operand order checked by reading, not run on hardware
//
#include <hip/hip_runtime.h>

// ---------------------------------------------------------------------------
// Problem constants (from reference): B=64, S=512, I=512, H4=1024
// ---------------------------------------------------------------------------
#define BB 64
#define SS 512
#define II 512
#define HH 1024   // 4*hidden (gate dimension)

typedef __attribute__((ext_vector_type(16))) __bf16 v16bf;
typedef __attribute__((ext_vector_type(8)))  __bf16 v8bf;
typedef __attribute__((ext_vector_type(8)))  float  v8f;

// fp32 -> bf16 bits, round-to-nearest-even
__device__ __forceinline__ unsigned short f2bf_bits(float f) {
    unsigned u = __builtin_bit_cast(unsigned, f);
    u += 0x7FFFu + ((u >> 16) & 1u);
    return (unsigned short)(u >> 16);
}

__global__ __launch_bounds__(256) void cvt_f2bf_kernel(
    const float* __restrict__ in, unsigned short* __restrict__ out, int n) {
    int i = blockIdx.x * blockDim.x + threadIdx.x;
    if (i < n) out[i] = f2bf_bits(in[i]);
}

__global__ __launch_bounds__(256) void copy_f_kernel(
    const float* __restrict__ in, float* __restrict__ out, int n) {
    int i = blockIdx.x * blockDim.x + threadIdx.x;
    if (i < n) out[i] = in[i];
}

// Build a v16bf A/B fragment: per CDNA5 ISA, lane half 0 holds K = kb+{0..7, 16..23},
// lane half 1 holds K = kb+{8..15, 24..31}; both are two contiguous 16B runs.
__device__ __forceinline__ v16bf load_frag(const __bf16* __restrict__ row, int kb, int half) {
    int c0 = kb + half * 8;
    v8bf lo = *(const v8bf*)(row + c0);
    v8bf hi = *(const v8bf*)(row + c0 + 16);
    return __builtin_shufflevector(lo, hi, 0,1,2,3,4,5,6,7,8,9,10,11,12,13,14,15);
}

// ---------------------------------------------------------------------------
// Kernel 1: xz[m, g] = sum_k x[b, s, k] * Wih[g, k] + bih[g]   (m = s*B + b)
// One wave computes a 32x32 tile of the (32768 x 1024) output: 2x2 register
// tiling -> 4 WMMAs per k-step with each fragment reused twice (2:1 vmem:wmma).
// ---------------------------------------------------------------------------
__global__ __launch_bounds__(256) void xz_gemm_kernel(
    const unsigned short* __restrict__ xbf_u,
    const unsigned short* __restrict__ wbf_u,
    const float* __restrict__ bih,
    float* __restrict__ xz)
{
    const __bf16* xbf = reinterpret_cast<const __bf16*>(xbf_u);
    const __bf16* wbf = reinterpret_cast<const __bf16*>(wbf_u);

    const int lane  = threadIdx.x & 31;
    const int wave  = threadIdx.x >> 5;
    const int tile  = blockIdx.x * 8 + wave;   // 32768 tiles total
    const int tileM = tile >> 5;               // 1024 M-tiles of 32 rows
    const int tileN = tile & 31;               // 32 N-tiles of 32 cols
    const int half  = lane >> 4;
    const int r     = lane & 15;

    // A rows owned by this lane for the two M subtiles
    const int m0 = tileM * 32 + r;
    const int m1 = m0 + 16;
    const int b0 = m0 & (BB - 1), s0 = m0 >> 6;
    const int b1 = m1 & (BB - 1), s1 = m1 >> 6;
    // B columns owned by this lane for the two N subtiles
    const int n0 = tileN * 32 + r;
    const int n1 = n0 + 16;

    const __bf16* arow0 = xbf + ((size_t)b0 * SS + s0) * II;  // x[b,s,:] contiguous in K
    const __bf16* arow1 = xbf + ((size_t)b1 * SS + s1) * II;
    const __bf16* brow0 = wbf + (size_t)n0 * II;              // Wih[g,:] contiguous in K
    const __bf16* brow1 = wbf + (size_t)n1 * II;

    v8f acc00 = {}, acc01 = {}, acc10 = {}, acc11 = {};
    #pragma unroll 2
    for (int k = 0; k < II; k += 32) {
        v16bf a0 = load_frag(arow0, k, half);
        v16bf a1 = load_frag(arow1, k, half);
        v16bf bm0 = load_frag(brow0, k, half);
        v16bf bm1 = load_frag(brow1, k, half);
        acc00 = __builtin_amdgcn_wmma_f32_16x16x32_bf16(false, a0, false, bm0, (short)0, acc00, false, false);
        acc01 = __builtin_amdgcn_wmma_f32_16x16x32_bf16(false, a0, false, bm1, (short)0, acc01, false, false);
        acc10 = __builtin_amdgcn_wmma_f32_16x16x32_bf16(false, a1, false, bm0, (short)0, acc10, false, false);
        acc11 = __builtin_amdgcn_wmma_f32_16x16x32_bf16(false, a1, false, bm1, (short)0, acc11, false, false);
    }

    // C/D layout: VGPR i -> (M = i + 8*half, N = lane&15) within each 16x16 tile
    const int g0 = tileN * 32 + r;
    const int g1 = g0 + 16;
    const float bg0 = bih[g0];
    const float bg1 = bih[g1];
    #pragma unroll
    for (int i = 0; i < 8; ++i) {
        const int mr0 = tileM * 32 + i + half * 8;
        const int mr1 = mr0 + 16;
        xz[(size_t)mr0 * HH + g0] = acc00[i] + bg0;
        xz[(size_t)mr0 * HH + g1] = acc01[i] + bg1;
        xz[(size_t)mr1 * HH + g0] = acc10[i] + bg0;
        xz[(size_t)mr1 * HH + g1] = acc11[i] + bg1;
    }
}

// ---------------------------------------------------------------------------
// Kernel 2: one recurrent step.
//   z      = xz[s] + h @ Whh^T + bhh        (64 x 1024)
//   sg     = sigmoid(z); th = tanh(z)
//   c1     = sg*c + sg*th        (carried cell state)
//   c2     = sg*c1 + sg*th
//   h_new  = sg*tanh(c2)         (carried hidden + emitted to hidden_seq^T)
// 256 output tiles of 16x16; K=1024 is split 4 ways (4 cooperating waves per
// tile, 8 k-iterations each) with an LDS v8f reduction -> 1024 waves per step
// and a 4x shorter dependent K chain on the scan's critical path.
// ---------------------------------------------------------------------------
__global__ __launch_bounds__(256) void lstm_step_kernel(
    const float* __restrict__ xz,
    const unsigned short* __restrict__ whh_u,
    const float* __restrict__ bhh,
    const unsigned short* __restrict__ hin_u,
    unsigned short* __restrict__ hout_u,
    float* __restrict__ cbuf,
    float* __restrict__ out_hidden,
    float* __restrict__ out_h,
    float* __restrict__ out_c,
    int s)
{
    const __bf16* whhbf = reinterpret_cast<const __bf16*>(whh_u);
    const __bf16* hin   = reinterpret_cast<const __bf16*>(hin_u);

    const int lane  = threadIdx.x & 31;
    const int wave  = threadIdx.x >> 5;
    const int tileL = wave >> 2;               // 2 tiles per block
    const int kk    = wave & 3;                // K slice 0..3
    const int tile  = blockIdx.x * 2 + tileL;  // 256 tiles total
    const int tileM = tile >> 6;               // 4 M-tiles (B=64)
    const int tileN = tile & 63;               // 64 N-tiles (HH=1024)
    const int half  = lane >> 4;
    const int r     = lane & 15;

    const __bf16* arow = hin   + (size_t)(tileM * 16 + r) * HH;  // h[b,:]
    const __bf16* brow = whhbf + (size_t)(tileN * 16 + r) * HH;  // Whh[g,:]

    const int kbeg = kk * (HH / 4);            // 256-wide K slice per wave

    v8f acc = {};
    #pragma unroll 4
    for (int k = kbeg; k < kbeg + HH / 4; k += 32) {
        v16bf a = load_frag(arow, k, half);
        v16bf bm = load_frag(brow, k, half);
        acc = __builtin_amdgcn_wmma_f32_16x16x32_bf16(
            false, a, false, bm, (short)0, acc, false, false);
    }

    // Reduce the 4 K-slice partials through LDS (slices 1..3 -> slice 0).
    __shared__ v8f red[2][3][32];              // [tileL][slice-1][lane] = 6 KB
    if (kk != 0) red[tileL][kk - 1][lane] = acc;
    __syncthreads();
    if (kk != 0) return;
    #pragma unroll
    for (int p = 0; p < 3; ++p) {
        v8f t = red[tileL][p][lane];
        #pragma unroll
        for (int i = 0; i < 8; ++i) acc[i] += t[i];
    }

    const int g = tileN * 16 + r;
    const float bg = bhh[g];
    const float* xz_s = xz + (size_t)s * BB * HH;   // xz[s, :, :]

    #pragma unroll
    for (int i = 0; i < 8; ++i) {
        const int b = tileM * 16 + i + half * 8;
        const size_t idx = (size_t)b * HH + g;

        float z  = acc[i] + xz_s[idx] + bg;
        float sg = 1.0f / (1.0f + __expf(-z));
        float th = tanhf(z);

        float c_old = cbuf[idx];
        float c1 = sg * c_old + sg * th;
        float c2 = sg * c1 + sg * th;
        float hn = sg * tanhf(c2);

        cbuf[idx]   = c1;                // carried cell state
        hout_u[idx] = f2bf_bits(hn);     // carried hidden (bf16 for next step's A)

        // hidden_seq is (HH, S*B): element [g, s*B + b]
        out_hidden[(size_t)g * (SS * BB) + (size_t)s * BB + b] = hn;
        if (s == SS - 1) { out_h[idx] = hn; out_c[idx] = c1; }
    }
}

// ---------------------------------------------------------------------------
// Host launcher
// ---------------------------------------------------------------------------
extern "C" void kernel_launch(void* const* d_in, const int* in_sizes, int n_in,
                              void* d_out, int out_size, void* d_ws, size_t ws_size,
                              hipStream_t stream) {
    const float* x   = (const float*)d_in[0];   // (B, S, I)
    const float* h0  = (const float*)d_in[1];   // (B, HH)
    const float* c0  = (const float*)d_in[2];   // (B, HH)
    const float* Wih = (const float*)d_in[3];   // (HH, I)
    const float* bih = (const float*)d_in[4];   // (HH,)
    const float* Whh = (const float*)d_in[5];   // (HH, HH)
    const float* bhh = (const float*)d_in[6];   // (HH,)

    // Workspace carve-out (all offsets 16B-aligned)
    char* ws = (char*)d_ws;
    size_t off = 0;
    float* xz = (float*)(ws + off);               off += (size_t)SS * BB * HH * sizeof(float);   // 128 MB
    unsigned short* xbf   = (unsigned short*)(ws + off); off += (size_t)BB * SS * II * 2;        // 32 MB
    unsigned short* wihbf = (unsigned short*)(ws + off); off += (size_t)HH * II * 2;             // 1 MB
    unsigned short* whhbf = (unsigned short*)(ws + off); off += (size_t)HH * HH * 2;             // 2 MB
    unsigned short* hbf0  = (unsigned short*)(ws + off); off += (size_t)BB * HH * 2;
    unsigned short* hbf1  = (unsigned short*)(ws + off); off += (size_t)BB * HH * 2;
    float* cbuf = (float*)(ws + off);             off += (size_t)BB * HH * sizeof(float);

    // Output layout: hidden_seq (HH x S*B) fp32, then final h, then final c
    float* out_hidden = (float*)d_out;
    float* out_h = out_hidden + (size_t)HH * SS * BB;
    float* out_c = out_h + (size_t)BB * HH;

    const int nx  = BB * SS * II;   // 16,777,216
    const int nwi = HH * II;        // 524,288
    const int nwh = HH * HH;        // 1,048,576
    const int nh  = BB * HH;        // 65,536

    cvt_f2bf_kernel<<<(nx  + 255) / 256, 256, 0, stream>>>(x,   xbf,   nx);
    cvt_f2bf_kernel<<<(nwi + 255) / 256, 256, 0, stream>>>(Wih, wihbf, nwi);
    cvt_f2bf_kernel<<<(nwh + 255) / 256, 256, 0, stream>>>(Whh, whhbf, nwh);
    cvt_f2bf_kernel<<<(nh  + 255) / 256, 256, 0, stream>>>(h0,  hbf0,  nh);
    copy_f_kernel  <<<(nh  + 255) / 256, 256, 0, stream>>>(c0,  cbuf,  nh);

    // Pre-projection GEMM: 1024 x 32 tiles of 32x32, 8 waves/block -> 4096 blocks
    xz_gemm_kernel<<<4096, 256, 0, stream>>>(xbf, wihbf, bih, xz);

    // Sequential recurrence: 256 tiles x 4 K-slices = 1024 waves -> 128 blocks/step
    for (int s = 0; s < SS; ++s) {
        const unsigned short* hin = (s & 1) ? hbf1 : hbf0;
        unsigned short*       hout = (s & 1) ? hbf0 : hbf1;
        lstm_step_kernel<<<128, 256, 0, stream>>>(xz, whhbf, bhh, hin, hout, cbuf,
                                                  out_hidden, out_h, out_c, s);
    }
}
